// NMRSetEncoder_88347477278885
// MI455X (gfx1250) — matrix operand order repaired and checked by reading, not compiled, not run on hardware
//
#include <hip/hip_runtime.h>

typedef unsigned short u16;
typedef __attribute__((ext_vector_type(16))) __bf16 v16bf;
typedef __attribute__((ext_vector_type(8)))  float  v8f;

union ABu { uint4 q[2]; v16bf v; __bf16 h[16]; u16 s[16]; };

__device__ inline u16 f2bf_bits(float f) {
  __bf16 h = (__bf16)f;
  return __builtin_bit_cast(u16, h);
}

// 16-lane butterfly reductions via v_permlane16_b32 (VALU, not DS)
__device__ inline float plx(float v, unsigned s0, unsigned s1) {
  unsigned u = __builtin_bit_cast(unsigned, v);
  u = __builtin_amdgcn_permlane16(u, u, s0, s1, false, false);
  return __builtin_bit_cast(float, u);
}
__device__ inline float rowmax16(float t) {
  t = fmaxf(t, plx(t, 0x67452301u, 0xEFCDAB89u));  // xor 1
  t = fmaxf(t, plx(t, 0x54761032u, 0xDCFE98BAu));  // xor 2
  t = fmaxf(t, plx(t, 0x32107654u, 0xBA98FEDCu));  // xor 4
  t = fmaxf(t, plx(t, 0xFEDCBA98u, 0x76543210u));  // xor 8
  return t;
}
__device__ inline float rowsum16(float t) {
  t += plx(t, 0x67452301u, 0xEFCDAB89u);
  t += plx(t, 0x54761032u, 0xDCFE98BAu);
  t += plx(t, 0x32107654u, 0xBA98FEDCu);
  t += plx(t, 0xFEDCBA98u, 0x76543210u);
  return t;
}

// ---------------------------------------------------------------------------
#define Bc   8
#define Lt   768
#define Dm   512
#define NHh  8
#define DHh  64
#define DFFm 2048
#define Mrows (Bc*Lt)   // 6144
#define ASTR 40         // padded LDS row stride (u16 units) for bank-freedom

// ---------------------------------------------------------------------------
// WMMA GEMM: out = A @ W^T + bias.
//   A: (M x K) row-major, f32 (AMODE=0, sync cvt staging) or bf16
//      (AMODE=1, global_load_async_to_lds_b64 staging)
//   W: (N x K) row-major f32 (cvt once per wave, reused by 4 WMMAs)
//   OMODE: 0 f32, 1 bf16, 2 bf16+relu, 3 bf16 V-transpose
// Block: 256 thr = 8 waves as 4(Mquad) x 2(N); block tile 256 x 32.
// Each wave: 64x16 strip = 4 WMMAs per K-step sharing one B operand.
// Requires K%32==0, M%256==0, N%32==0.
// ---------------------------------------------------------------------------
template <int AMODE, int OMODE>
__global__ __launch_bounds__(256) void gemm_kernel(
    const void* __restrict__ Ap, int lda,
    const float* __restrict__ W, const float* __restrict__ bias,
    void* __restrict__ outp, int ldo, int col0,
    int M, int N, int K) {
  __shared__ __align__(16) u16 At[256 * ASTR];
  const int tid = threadIdx.x;
  const int wid = tid >> 5, lane = tid & 31;
  const int h2 = lane >> 4, nn = lane & 15;
  const int ni = wid & 1, mq = wid >> 1;
  const int m0 = blockIdx.y * 256;
  const int n0 = blockIdx.x * 32;
  const int ng = n0 + ni * 16 + nn;
  const unsigned lds_base = (unsigned)(size_t)(void*)At;  // low 32 = LDS offset

  v8f acc[4] = {{}, {}, {}, {}};
  for (int k = 0; k < K; k += 32) {
    __syncthreads();
    // stage A tile 256x32 (bf16) into LDS
    if constexpr (AMODE == 0) {
#pragma unroll
      for (int i = 0; i < 8; ++i) {
        const int c = tid + 256 * i;
        const int row = c >> 3, kk = (c & 7) * 4;
        const float4 f = *(const float4*)((const float*)Ap +
                                          (size_t)(m0 + row) * lda + k + kk);
        uint2 p;
        p.x = (unsigned)f2bf_bits(f.x) | ((unsigned)f2bf_bits(f.y) << 16);
        p.y = (unsigned)f2bf_bits(f.z) | ((unsigned)f2bf_bits(f.w) << 16);
        *(uint2*)&At[row * ASTR + kk] = p;
      }
    } else {
#pragma unroll
      for (int i = 0; i < 8; ++i) {
        const int c = tid + 256 * i;
        const int row = c >> 3, kk = (c & 7) * 4;
        const unsigned lo = lds_base + (unsigned)(row * ASTR + kk) * 2;
        const u16* gp = (const u16*)Ap + (size_t)(m0 + row) * lda + k + kk;
        asm volatile("global_load_async_to_lds_b64 %0, %1, off"
                     :: "v"(lo), "v"(gp) : "memory");
      }
      asm volatile("s_wait_asynccnt 0" ::: "memory");
    }
    __syncthreads();
    // B operand: 16 contiguous f32 of W row, cvt to bf16; reused 4x
    ABu bm;
    const float* wp = W + (size_t)ng * K + k + 16 * h2;
#pragma unroll
    for (int e = 0; e < 16; ++e) bm.h[e] = (__bf16)wp[e];
    if (k + 32 < K) __builtin_prefetch(wp + 32, 0, 0);  // global_prefetch
#pragma unroll
    for (int mt = 0; mt < 4; ++mt) {
      ABu a;
      const int row = mq * 64 + mt * 16 + nn;
      a.q[0] = *(const uint4*)&At[row * ASTR + 16 * h2];
      a.q[1] = *(const uint4*)&At[row * ASTR + 8 + 16 * h2];
      acc[mt] = __builtin_amdgcn_wmma_f32_16x16x32_bf16(
          false, a.v, false, bm.v, (short)0, acc[mt], false, false);
    }
  }
  const float bv = bias ? bias[ng] : 0.f;
#pragma unroll
  for (int mt = 0; mt < 4; ++mt)
#pragma unroll
    for (int r = 0; r < 8; ++r) {
      const float v = acc[mt][r] + bv;
      const int mg = m0 + mq * 64 + mt * 16 + r + 8 * h2;
      if (OMODE == 0) {
        ((float*)outp)[(size_t)mg * ldo + col0 + ng] = v;
      } else if (OMODE == 1) {
        ((__bf16*)outp)[(size_t)mg * ldo + col0 + ng] = (__bf16)v;
      } else if (OMODE == 2) {
        ((__bf16*)outp)[(size_t)mg * ldo + col0 + ng] = (__bf16)fmaxf(v, 0.f);
      } else {  // Vt[((b*NH+h)*DH+dh)*L + l]
        const int bb = mg / Lt, l = mg % Lt;
        const int hh = ng >> 6, dh = ng & 63;
        ((__bf16*)outp)[(size_t)(((bb * NHh + hh) << 6) + dh) * Lt + l] =
            (__bf16)v;
      }
    }
}

// ---------------------------------------------------------------------------
// Fused flash attention: one wave per (b, h, 16-row q tile).
// Q,K bf16 row-major; Vt bf16 transposed. Output bf16 row-major (Mrows x D).
// ---------------------------------------------------------------------------
__global__ __launch_bounds__(256) void attn_kernel(
    const u16* __restrict__ Qb, const u16* __restrict__ Kb,
    const u16* __restrict__ Vt, const float* __restrict__ bias,
    u16* __restrict__ outp) {
  __shared__ __align__(16) u16 Pt[8 * 16 * 32];
  const int tid = threadIdx.x;
  const int wid = tid >> 5, lane = tid & 31;
  const int h2 = lane >> 4, nn = lane & 15;
  const int gw = blockIdx.x * 8 + wid;
  const int b = gw / (NHh * (Lt / 16));
  const int rem = gw % (NHh * (Lt / 16));
  const int h = rem / (Lt / 16), qt = rem % (Lt / 16);
  const int q0 = qt * 16;
  u16* P = &Pt[wid * 512];

  ABu aq0, aq1;
  {
    const size_t base = (size_t)(b * Lt + q0 + nn) * Dm + h * DHh;
    aq0.q[0] = *(const uint4*)&Qb[base + 16 * h2];
    aq0.q[1] = *(const uint4*)&Qb[base + 8 + 16 * h2];
    aq1.q[0] = *(const uint4*)&Qb[base + 32 + 16 * h2];
    aq1.q[1] = *(const uint4*)&Qb[base + 40 + 16 * h2];
  }
  float Mr[8], Sr[8];
  v8f accv[4] = {{}, {}, {}, {}};
#pragma unroll
  for (int r = 0; r < 8; ++r) { Mr[r] = -3.0e38f; Sr[r] = 0.f; }

  for (int kb = 0; kb < Lt; kb += 32) {
    v8f s[2];
#pragma unroll
    for (int t = 0; t < 2; ++t) {
      const size_t krow = (size_t)(b * Lt + kb + t * 16 + nn) * Dm + h * DHh;
      const v16bf bk0 = *(const v16bf*)&Kb[krow + 16 * h2];
      const v16bf bk1 = *(const v16bf*)&Kb[krow + 32 + 16 * h2];
      v8f c = {};
      c = __builtin_amdgcn_wmma_f32_16x16x32_bf16(false, aq0.v, false, bk0,
                                                  (short)0, c, false, false);
      c = __builtin_amdgcn_wmma_f32_16x16x32_bf16(false, aq1.v, false, bk1,
                                                  (short)0, c, false, false);
#pragma unroll
      for (int r = 0; r < 8; ++r) {
        c[r] = c[r] * 0.125f +
               bias[(size_t)(b * Lt + q0 + r + 8 * h2) * Lt + kb + t * 16 + nn];
      }
      s[t] = c;
    }
#pragma unroll
    for (int r = 0; r < 8; ++r) {
      const float t0 = rowmax16(fmaxf(s[0][r], s[1][r]));
      const float mnew = fmaxf(Mr[r], t0);
      const float alpha = __expf(Mr[r] - mnew);
      const float p0 = __expf(s[0][r] - mnew);
      const float p1 = __expf(s[1][r] - mnew);
      const float rs = rowsum16(p0 + p1);
      Sr[r] = Sr[r] * alpha + rs;
      Mr[r] = mnew;
#pragma unroll
      for (int nt = 0; nt < 4; ++nt) accv[nt][r] *= alpha;
      P[(r + 8 * h2) * 32 + nn] = f2bf_bits(p0);
      P[(r + 8 * h2) * 32 + 16 + nn] = f2bf_bits(p1);
    }
    asm volatile("s_wait_dscnt 0" ::: "memory");
    ABu ap;
    ap.q[0] = *(const uint4*)&P[nn * 32 + 16 * h2];
    ap.q[1] = *(const uint4*)&P[nn * 32 + 8 + 16 * h2];
#pragma unroll
    for (int nt = 0; nt < 4; ++nt) {
      const size_t vrow =
          (size_t)((b * NHh + h) * DHh + nt * 16 + nn) * Lt + kb + 16 * h2;
      const v16bf bv = *(const v16bf*)&Vt[vrow];
      accv[nt] = __builtin_amdgcn_wmma_f32_16x16x32_bf16(
          false, ap.v, false, bv, (short)0, accv[nt], false, false);
    }
  }
#pragma unroll
  for (int nt = 0; nt < 4; ++nt)
#pragma unroll
    for (int r = 0; r < 8; ++r) {
      const int mg = b * Lt + q0 + r + 8 * h2;
      outp[(size_t)mg * Dm + h * DHh + nt * 16 + nn] =
          f2bf_bits(accv[nt][r] / Sr[r]);
    }
}

// ---------------------------------------------------------------------------
// x = LayerNorm(x + g); also writes bf16 mirror xb
// ---------------------------------------------------------------------------
__global__ __launch_bounds__(256) void add_ln_kernel(
    float* __restrict__ x, u16* __restrict__ xb, const float* __restrict__ g,
    const float* __restrict__ gamma, const float* __restrict__ beta) {
  __shared__ float red[256];
  const int tid = threadIdx.x;
  const size_t base = (size_t)blockIdx.x * Dm;
  const float v0 = x[base + tid] + g[base + tid];
  const float v1 = x[base + 256 + tid] + g[base + 256 + tid];
  red[tid] = v0 + v1;
  __syncthreads();
  for (int s = 128; s > 0; s >>= 1) {
    if (tid < s) red[tid] += red[tid + s];
    __syncthreads();
  }
  const float mean = red[0] * (1.f / 512.f);
  __syncthreads();
  const float d0 = v0 - mean, d1 = v1 - mean;
  red[tid] = d0 * d0 + d1 * d1;
  __syncthreads();
  for (int s = 128; s > 0; s >>= 1) {
    if (tid < s) red[tid] += red[tid + s];
    __syncthreads();
  }
  const float rstd = rsqrtf(red[0] * (1.f / 512.f) + 1e-5f);
  const float o0 = d0 * rstd * gamma[tid] + beta[tid];
  const float o1 = d1 * rstd * gamma[256 + tid] + beta[256 + tid];
  x[base + tid] = o0;
  x[base + 256 + tid] = o1;
  xb[base + tid] = f2bf_bits(o0);
  xb[base + 256 + tid] = f2bf_bits(o1);
}

// ---------------------------------------------------------------------------
__global__ __launch_bounds__(256) void encode_phi_kernel(
    const float* __restrict__ hsqc, const float* __restrict__ h_nmr,
    const float* __restrict__ c_nmr, const float* __restrict__ cH,
    const float* __restrict__ lsH, const float* __restrict__ cC,
    const float* __restrict__ lsC, float* __restrict__ phiH,
    float* __restrict__ phiC, float* __restrict__ Hall,
    float* __restrict__ Call, float* __restrict__ mH, float* __restrict__ mC) {
  const int idx = blockIdx.x * 256 + threadIdx.x;  // Mrows*64
  const int i = idx >> 6, j = idx & 63;
  const int b = i / Lt, l = i % Lt;
  float H, C;
  if (l < 256) {
    H = hsqc[(b * 256 + l) * 2 + 0];
    C = hsqc[(b * 256 + l) * 2 + 1];
  } else if (l < 512) {
    H = h_nmr[(b * 256 + l - 256) * 3 + 1];
    C = 0.f;
  } else {
    H = 0.f;
    C = c_nmr[(b * 256 + l - 512) * 3 + 0];
  }
  float sig = fmaxf(__expf(lsH[j]), 1e-4f);
  float d = H - cH[j];
  phiH[(size_t)i * 64 + j] = __expf(-d * d / (2.f * sig * sig));
  sig = fmaxf(__expf(lsC[j]), 1e-4f);
  d = C - cC[j];
  phiC[(size_t)i * 64 + j] = __expf(-d * d / (2.f * sig * sig));
  if (j == 0) {
    Hall[i] = H;
    Call[i] = C;
    mH[i] = (fabsf(H) > 0.f) ? 1.f : 0.f;
    mC[i] = (fabsf(C) > 0.f) ? 1.f : 0.f;
  }
}

__global__ __launch_bounds__(256) void mixmask_kernel(
    float* __restrict__ feat, const float* __restrict__ mH,
    const float* __restrict__ mC, const float* __restrict__ missH,
    const float* __restrict__ missC, const float* __restrict__ type_emb) {
  const int idx = blockIdx.x * 256 + threadIdx.x;  // Mrows*544
  const int i = idx / 544, c = idx % 544;
  float* p = &feat[(size_t)i * 544 + c];
  if (c < 256) {
    const float a = mH[i];
    *p = *p * a + missH[c] * (1.f - a);
  } else if (c < 512) {
    const float a = mC[i];
    *p = *p * a + missC[c - 256] * (1.f - a);
  } else {
    const int l = i % Lt;
    const int t = (l < 256) ? 2 : ((l < 512) ? 0 : 1);
    *p = type_emb[t * 32 + (c - 512)];
  }
}

__global__ __launch_bounds__(256) void pair_bias_kernel(
    const float* __restrict__ Hall, const float* __restrict__ Call,
    const float* __restrict__ mH, const float* __restrict__ mC,
    const float* __restrict__ cH, const float* __restrict__ lsH,
    const float* __restrict__ wH, const float* __restrict__ scH,
    const float* __restrict__ cC, const float* __restrict__ lsC,
    const float* __restrict__ wC, const float* __restrict__ scC,
    float* __restrict__ bias) {
  __shared__ float sc[32], si[32], sw[32], tc[32], ti[32], tw[32];
  const int tid = threadIdx.x;
  if (tid < 32) {
    sc[tid] = cH[tid];
    const float s = fmaxf(__expf(lsH[tid]), 1e-4f);
    si[tid] = 1.f / (2.f * s * s);
    sw[tid] = wH[tid];
  } else if (tid < 64) {
    const int t = tid - 32;
    tc[t] = cC[t];
    const float s = fmaxf(__expf(lsC[t]), 1e-4f);
    ti[t] = 1.f / (2.f * s * s);
    tw[t] = wC[t];
  }
  __syncthreads();
  const size_t idx = (size_t)blockIdx.x * 256 + tid;  // B*L*L
  const int b = (int)(idx / (Lt * Lt));
  const int r = (int)(idx % (Lt * Lt));
  const int i = r / Lt, j = r % Lt;
  const float dH = fabsf(Hall[b * Lt + i] - Hall[b * Lt + j]);
  const float dC = fabsf(Call[b * Lt + i] - Call[b * Lt + j]);
  float sH = 0.f, sC2 = 0.f;
#pragma unroll 8
  for (int p = 0; p < 32; ++p) {
    float d = dH - sc[p];
    sH += __expf(-d * d * si[p]) * sw[p];
    d = dC - tc[p];
    sC2 += __expf(-d * d * ti[p]) * tw[p];
  }
  const float m1 = (mH[b * Lt + i] > 0.5f && mH[b * Lt + j] > 0.5f) ? 1.f : 0.f;
  const float m2 = (mC[b * Lt + i] > 0.5f && mC[b * Lt + j] > 0.5f) ? 1.f : 0.f;
  bias[idx] = scH[0] * sH * m1 + scC[0] * sC2 * m2;
}

__global__ __launch_bounds__(256) void film_kernel(
    const float* __restrict__ mw, const float* __restrict__ W1,
    const float* __restrict__ b1, const float* __restrict__ W2,
    const float* __restrict__ b2, float* __restrict__ gb) {
  __shared__ float hid[64];
  const int b = blockIdx.x, tid = threadIdx.x;
  const float t = mw[b];
  if (tid < 64) hid[tid] = fmaxf(W1[tid] * t + b1[tid], 0.f);
  __syncthreads();
  for (int o = tid; o < 1024; o += 256) {
    float s = b2[o];
#pragma unroll 8
    for (int j = 0; j < 64; ++j) s += W2[o * 64 + j] * hid[j];
    gb[b * 1024 + o] = s;
  }
}

__global__ __launch_bounds__(256) void film_apply_kernel(
    float* __restrict__ x, u16* __restrict__ xb, const float* __restrict__ gb) {
  const size_t idx = (size_t)blockIdx.x * 256 + threadIdx.x;  // Mrows*D
  const int m = (int)(idx >> 9), d = (int)(idx & 511);
  const int b = m / Lt;
  const float v = x[idx] * (1.f + gb[b * 1024 + d]) + gb[b * 1024 + 512 + d];
  x[idx] = v;
  xb[idx] = f2bf_bits(v);
}

// ---------------------------------------------------------------------------
extern "C" void kernel_launch(void* const* d_in, const int* in_sizes, int n_in,
                              void* d_out, int out_size, void* d_ws,
                              size_t ws_size, hipStream_t stream) {
  (void)in_sizes; (void)n_in; (void)out_size; (void)ws_size;
  auto F = [&](int i) { return (const float*)d_in[i]; };
  float* x = (float*)d_out;  // activations live in d_out

  char* ws = (char*)d_ws;
  size_t off = 0;
  auto alloc = [&](size_t bytes) -> char* {
    char* p = ws + off;
    off += (bytes + 255) & ~(size_t)255;
    return p;
  };
  float* bias = (float*)alloc((size_t)Bc * Lt * Lt * 4);
  u16* xb = (u16*)alloc((size_t)Mrows * Dm * 2);
  u16* Qb = (u16*)alloc((size_t)Mrows * Dm * 2);
  u16* Kb = (u16*)alloc((size_t)Mrows * Dm * 2);
  u16* Vt = (u16*)alloc((size_t)Mrows * Dm * 2);
  u16* attnb = (u16*)alloc((size_t)Mrows * Dm * 2);
  float* gbuf = (float*)alloc((size_t)Mrows * Dm * 4);
  char* region = alloc((size_t)Mrows * DFFm * 2);  // FFN hidden (bf16)
  float* feat = (float*)region;                    // reused during encode
  u16* hidden = (u16*)region;
  float* phiH = (float*)alloc((size_t)Mrows * 64 * 4);
  float* phiC = (float*)alloc((size_t)Mrows * 64 * 4);
  float* Hall = (float*)alloc(Mrows * 4);
  float* Call = (float*)alloc(Mrows * 4);
  float* mHb = (float*)alloc(Mrows * 4);
  float* mCb = (float*)alloc(Mrows * 4);
  float* filmgb = (float*)alloc(Bc * 1024 * 4);

  // ---- encode front-end -------------------------------------------------
  encode_phi_kernel<<<Mrows * 64 / 256, 256, 0, stream>>>(
      F(0), F(1), F(2), F(4), F(5), F(8), F(9), phiH, phiC, Hall, Call, mHb,
      mCb);
  gemm_kernel<0, 0><<<dim3(256 / 32, Mrows / 256), 256, 0, stream>>>(
      phiH, 64, F(6), F(7), feat, 544, 0, Mrows, 256, 64);
  gemm_kernel<0, 0><<<dim3(256 / 32, Mrows / 256), 256, 0, stream>>>(
      phiC, 64, F(10), F(11), feat, 544, 256, Mrows, 256, 64);
  mixmask_kernel<<<Mrows * 544 / 256, 256, 0, stream>>>(feat, mHb, mCb, F(12),
                                                        F(13), F(14));
  gemm_kernel<0, 0><<<dim3(Dm / 32, Mrows / 256), 256, 0, stream>>>(
      feat, 544, F(15), F(16), x, Dm, 0, Mrows, Dm, 544);
  pair_bias_kernel<<<Bc * Lt * Lt / 256, 256, 0, stream>>>(
      Hall, Call, mHb, mCb, F(17), F(18), F(19), F(20), F(21), F(22), F(23),
      F(24), bias);

  // ---- transformer layers ----------------------------------------------
  for (int l = 0; l < 6; ++l) {
    const int pb = 25 + 16 * l;   // block params base
    const int fb = 121 + 4 * l;   // film params base
    film_kernel<<<Bc, 256, 0, stream>>>(F(3), F(fb + 0), F(fb + 1), F(fb + 2),
                                        F(fb + 3), filmgb);
    film_apply_kernel<<<(size_t)Mrows * Dm / 256, 256, 0, stream>>>(x, xb,
                                                                    filmgb);
    gemm_kernel<1, 1><<<dim3(Dm / 32, Mrows / 256), 256, 0, stream>>>(
        xb, Dm, F(pb + 0), F(pb + 1), Qb, Dm, 0, Mrows, Dm, Dm);
    gemm_kernel<1, 1><<<dim3(Dm / 32, Mrows / 256), 256, 0, stream>>>(
        xb, Dm, F(pb + 2), F(pb + 3), Kb, Dm, 0, Mrows, Dm, Dm);
    gemm_kernel<1, 3><<<dim3(Dm / 32, Mrows / 256), 256, 0, stream>>>(
        xb, Dm, F(pb + 4), F(pb + 5), Vt, 0, 0, Mrows, Dm, Dm);
    attn_kernel<<<Bc * NHh * (Lt / 16) / 8, 256, 0, stream>>>(Qb, Kb, Vt, bias,
                                                              attnb);
    gemm_kernel<1, 0><<<dim3(Dm / 32, Mrows / 256), 256, 0, stream>>>(
        attnb, Dm, F(pb + 6), F(pb + 7), gbuf, Dm, 0, Mrows, Dm, Dm);
    add_ln_kernel<<<Mrows, 256, 0, stream>>>(x, xb, gbuf, F(pb + 8),
                                             F(pb + 9));
    gemm_kernel<1, 2><<<dim3(DFFm / 32, Mrows / 256), 256, 0, stream>>>(
        xb, Dm, F(pb + 10), F(pb + 11), hidden, DFFm, 0, Mrows, DFFm, Dm);
    gemm_kernel<1, 0><<<dim3(Dm / 32, Mrows / 256), 256, 0, stream>>>(
        hidden, DFFm, F(pb + 12), F(pb + 13), gbuf, Dm, 0, Mrows, Dm, DFFm);
    add_ln_kernel<<<Mrows, 256, 0, stream>>>(x, xb, gbuf, F(pb + 14),
                                             F(pb + 15));
  }
}
